// TELlamaDecoderLayer_18528488915272
// MI455X (gfx1250) — matrix operand order repaired and checked by reading, not compiled, not run on hardware
//
#include <hip/hip_runtime.h>
#include <math.h>

// ---------- types ----------
typedef __bf16 bf16;
typedef __attribute__((ext_vector_type(8)))  bf16  bf16x8;
typedef __attribute__((ext_vector_type(16))) bf16  v16bf;
typedef __attribute__((ext_vector_type(8)))  float v8f;
typedef __attribute__((ext_vector_type(4)))  float f32x4;

union FragBF16 { v16bf v; bf16x8 h[2]; };

static __device__ __forceinline__ v8f wmma_bf16(v16bf a, v16bf b, v8f c) {
  // D = A(16x32) * B(32x16) + C(16x16), fp32 accumulate
  return __builtin_amdgcn_wmma_f32_16x16x32_bf16(false, a, false, b, (short)0, c, false, false);
}

// ---------- model constants ----------
constexpr int BATCH = 2;
constexpr int SEQ = 2048;
constexpr int HID = 2048;
constexpr int NH = 16;
constexpr int NKV = 4;
constexpr int HD = 128;
constexpr int INTER = 5632;
constexpr int MTOK = BATCH * SEQ;                       // 4096 rows
constexpr float ATTN_SCALE = 0.08838834764831845f;      // 1/sqrt(128)
constexpr float LOG2E = 1.4426950408889634f;
constexpr float SCALE_LOG2E = ATTN_SCALE * LOG2E;       // exp2-domain scale

// =====================================================================
// RMSNorm: fp32 row -> bf16 row  (one 256-thread block per row, H=2048)
// =====================================================================
__global__ __launch_bounds__(256) void rmsnorm_kernel(const float* __restrict__ x,
                                                      const float* __restrict__ w,
                                                      bf16* __restrict__ out) {
  constexpr int H = HID;
  const int row = blockIdx.x;
  const int tid = threadIdx.x;
  const float* xr = x + (size_t)row * H;
  float ss = 0.f;
  for (int i = tid; i < H; i += 256) { float v = xr[i]; ss += v * v; }
  __shared__ float red[256];
  red[tid] = ss; __syncthreads();
  for (int s = 128; s > 0; s >>= 1) { if (tid < s) red[tid] += red[tid + s]; __syncthreads(); }
  const float rinv = rsqrtf(red[0] / (float)H + 1e-5f);
  bf16* orow = out + (size_t)row * H;
  for (int i = tid; i < H; i += 256) orow[i] = (bf16)(xr[i] * rinv * w[i]);
}

// =====================================================================
// GEMM: C[M,N] = A[M,K](bf16) @ W[N,K](fp32 -> bf16)^T
//   Double-buffered global->reg->LDS staging; 128x128x32 block tile,
//   8 waves (2x4), wave tile 64x32 (8 independent WMMA chains).
// =====================================================================
template <typename OutT, bool SWIGLU, bool RESID>
__global__ __launch_bounds__(256) void gemm_wmma_kernel(const bf16* __restrict__ A, int lda,
                                                        const float* __restrict__ W,  // [N,K]
                                                        OutT* __restrict__ C,
                                                        const float* __restrict__ resid,
                                                        int M, int N, int K) {
  constexpr int LDT = 40;  // 80 B row stride (16B multiple)
  __shared__ __align__(16) bf16 As[128][LDT];
  __shared__ __align__(16) bf16 Bs[128][LDT];

  const int tid  = threadIdx.x;
  const int lane = tid & 31;
  const int wave = tid >> 5;
  const int wrow = wave >> 2;   // 0..1  -> 64 rows each
  const int wcol = wave & 3;    // 0..3  -> 32 cols each
  const int tileM = blockIdx.y * 128;
  const int tileN = blockIdx.x * 128;

  v8f acc[4][2];
#pragma unroll
  for (int i = 0; i < 4; ++i)
#pragma unroll
    for (int j = 0; j < 2; ++j)
#pragma unroll
      for (int r = 0; r < 8; ++r) acc[i][j][r] = 0.f;

  const int ar = tid >> 1;            // A tile row this thread stages
  const int ac = (tid & 1) * 16;      // 16 halfs
  const int bn = tid >> 1;            // W tile row (output col)
  const int bc = (tid & 1) * 16;      // 16 floats

  const bf16* Arow = A + (size_t)(tileM + ar) * lda + ac;
  const float* Wrow = W + (size_t)(tileN + bn) * K + bc;

  auto loadA = [&](int k0, bf16x8& r0, bf16x8& r1) {
    if constexpr (!SWIGLU) {
      const bf16* ap = Arow + k0;
      r0 = *(const bf16x8*)(ap);
      r1 = *(const bf16x8*)(ap + 8);
    } else {
      const bf16* gp = Arow + k0;   // gate
      const bf16* up = gp + K;      // up
#pragma unroll
      for (int i = 0; i < 8; ++i) {
        float g = (float)gp[i], u = (float)up[i];
        r0[i] = (bf16)((g / (1.f + __expf(-g))) * u);
      }
#pragma unroll
      for (int i = 0; i < 8; ++i) {
        float g = (float)gp[8 + i], u = (float)up[8 + i];
        r1[i] = (bf16)((g / (1.f + __expf(-g))) * u);
      }
    }
  };
  auto loadW = [&](int k0, f32x4 wv[4]) {
#pragma unroll
    for (int c = 0; c < 4; ++c) wv[c] = *(const f32x4*)(Wrow + k0 + 4 * c);
  };

  bf16x8 a0, a1, na0, na1;
  f32x4 w4[4], nw4[4];
  loadA(0, a0, a1);
  loadW(0, w4);

  for (int k0 = 0; k0 < K; k0 += 32) {
    // ---- commit staged tile to LDS ----
    *(bf16x8*)&As[ar][ac]     = a0;
    *(bf16x8*)&As[ar][ac + 8] = a1;
#pragma unroll
    for (int c = 0; c < 4; ++c)
#pragma unroll
      for (int e = 0; e < 4; ++e) Bs[bn][bc + 4 * c + e] = (bf16)w4[c][e];
    __syncthreads();

    // ---- issue next tile's global loads (overlap with WMMA) ----
    if (k0 + 32 < K) { loadA(k0 + 32, na0, na1); loadW(k0 + 32, nw4); }

    // ---- preload all fragments, then 8 independent WMMA chains ----
    const int mh = lane & 15, kh = lane >> 4;
    FragBF16 afr[4], bfr[2];
#pragma unroll
    for (int i = 0; i < 4; ++i) {
      const int m = wrow * 64 + i * 16 + mh;
      afr[i].h[0] = *(const bf16x8*)&As[m][kh * 8];
      afr[i].h[1] = *(const bf16x8*)&As[m][16 + kh * 8];
    }
#pragma unroll
    for (int j = 0; j < 2; ++j) {
      const int n = wcol * 32 + j * 16 + mh;
      bfr[j].h[0] = *(const bf16x8*)&Bs[n][kh * 8];
      bfr[j].h[1] = *(const bf16x8*)&Bs[n][16 + kh * 8];
    }
#pragma unroll
    for (int i = 0; i < 4; ++i)
#pragma unroll
      for (int j = 0; j < 2; ++j)
        acc[i][j] = wmma_bf16(afr[i].v, bfr[j].v, acc[i][j]);
    __syncthreads();

    a0 = na0; a1 = na1;
#pragma unroll
    for (int c = 0; c < 4; ++c) w4[c] = nw4[c];
  }

  // ---- epilogue (C layout: lane half -> M+8, lane&15 -> N) ----
  const int rowsel = (lane >> 4) * 8;
  const int coln   = lane & 15;
#pragma unroll
  for (int i = 0; i < 4; ++i)
#pragma unroll
    for (int j = 0; j < 2; ++j)
#pragma unroll
      for (int r = 0; r < 8; ++r) {
        const int gm = tileM + wrow * 64 + i * 16 + rowsel + r;
        const int gn = tileN + wcol * 32 + j * 16 + coln;
        const size_t idx = (size_t)gm * N + gn;
        float v = acc[i][j][r];
        if (RESID) v += resid[idx];
        C[idx] = (OutT)v;
      }
}

// =====================================================================
// Flash attention (causal, GQA 16Q/4KV heads, hd=128, S=2048).
// Block = 256 thr (8 waves) handles 128 q rows of one (b, head).
// B-fragments are fully preloaded before each WMMA chain; the two
// 16-key score accumulators run as interleaved independent chains.
// =====================================================================
__global__ __launch_bounds__(256, 1) void attn_kernel(const bf16* __restrict__ q,
                                                      const bf16* __restrict__ k,
                                                      const bf16* __restrict__ v,
                                                      bf16* __restrict__ o) {
  __shared__ __align__(16) bf16 Ks[32][136];      // [key][hd], 272B stride
  __shared__ __align__(16) bf16 Vt[128][40];      // [hd][key], transposed
  __shared__ __align__(16) bf16 Pw[8][16][40];    // per-wave P relayout

  const int qb  = blockIdx.x;       // q block (128 rows)
  const int h   = blockIdx.y;       // head
  const int b   = blockIdx.z;       // batch
  const int kvh = h >> 2;           // GQA: kv head = h/4
  const int tid = threadIdx.x;
  const int lane = tid & 31;
  const int wave = tid >> 5;
  const int qbase = qb * 128 + wave * 16;
  const int mh = lane & 15, kh = lane >> 4;
  const int rowsel = kh * 8;
  const int qrow0 = qbase + rowsel;

  // q fragments for the wave's 16 rows, whole head_dim (4 chunks of 32)
  FragBF16 qf[4];
#pragma unroll
  for (int c = 0; c < 4; ++c) {
    const bf16* qp = q + ((size_t)(b * SEQ + qbase + mh)) * HID + h * HD + c * 32;
    qf[c].h[0] = *(const bf16x8*)(qp + kh * 8);
    qf[c].h[1] = *(const bf16x8*)(qp + 16 + kh * 8);
  }

  v8f oacc[8];
#pragma unroll
  for (int j = 0; j < 8; ++j)
#pragma unroll
    for (int r = 0; r < 8; ++r) oacc[j][r] = 0.f;
  float mrow[8], lrow[8];
#pragma unroll
  for (int r = 0; r < 8; ++r) { mrow[r] = -1e30f; lrow[r] = 0.f; }

  const bf16* kbase = k + (size_t)b * SEQ * (NKV * HD) + kvh * HD;
  const bf16* vbase = v + (size_t)b * SEQ * (NKV * HD) + kvh * HD;

  const int ntiles = (qb + 1) * 4;  // keys 0 .. qb*128+127
  for (int kt = 0; kt < ntiles; ++kt) {
    const int key0 = kt * 32;
    __syncthreads();
    // ---- cooperative K/V tile load (32 keys x 128 hd) ----
    {
      const int kr = tid >> 3;              // key 0..31
      const int kc = (tid & 7) * 16;        // hd chunk
      const bf16* ks = kbase + (size_t)(key0 + kr) * (NKV * HD) + kc;
      *(bf16x8*)&Ks[kr][kc]     = *(const bf16x8*)(ks);
      *(bf16x8*)&Ks[kr][kc + 8] = *(const bf16x8*)(ks + 8);
      const bf16* vs = vbase + (size_t)(key0 + kr) * (NKV * HD) + kc;
#pragma unroll
      for (int i = 0; i < 16; ++i) Vt[kc + i][kr] = vs[i];
      if (kt + 1 < ntiles) {   // prefetch next tile (global_prefetch_b8)
        __builtin_prefetch(ks + 32 * (NKV * HD), 0, 1);
        __builtin_prefetch(vs + 32 * (NKV * HD), 0, 1);
      }
    }
    __syncthreads();

    // wave-uniform tile classification
    const bool active = (key0 <= qbase + 15);     // any visible key
    const bool edge   = (key0 + 31 > qbase);      // straddles diagonal -> mask

    if (active) {
      // ---- preload all 8 K-fragments, then interleaved WMMA chains ----
      FragBF16 kf[2][4];
#pragma unroll
      for (int kh2 = 0; kh2 < 2; ++kh2) {
        const int n = kh2 * 16 + mh;        // key col
#pragma unroll
        for (int c = 0; c < 4; ++c) {
          kf[kh2][c].h[0] = *(const bf16x8*)&Ks[n][c * 32 + kh * 8];
          kf[kh2][c].h[1] = *(const bf16x8*)&Ks[n][c * 32 + 16 + kh * 8];
        }
      }
      v8f sc[2];
#pragma unroll
      for (int r = 0; r < 8; ++r) { sc[0][r] = 0.f; sc[1][r] = 0.f; }
#pragma unroll
      for (int c = 0; c < 4; ++c) {   // two independent accumulator chains
        sc[0] = wmma_bf16(qf[c].v, kf[0][c].v, sc[0]);
        sc[1] = wmma_bf16(qf[c].v, kf[1][c].v, sc[1]);
      }

      float sval[2][8];
#pragma unroll
      for (int kh2 = 0; kh2 < 2; ++kh2) {
        if (edge) {
          const int kgrel = key0 + kh2 * 16 + mh - qrow0;  // mask iff kgrel > r
#pragma unroll
          for (int r = 0; r < 8; ++r)
            sval[kh2][r] = (kgrel <= r) ? sc[kh2][r] * SCALE_LOG2E : -1e30f;
        } else {
#pragma unroll
          for (int r = 0; r < 8; ++r) sval[kh2][r] = sc[kh2][r] * SCALE_LOG2E;
        }
      }

      // ---- online softmax (exp2-domain; row stats via shfl_xor) ----
      float corr[8];
#pragma unroll
      for (int r = 0; r < 8; ++r) {
        float t = fmaxf(sval[0][r], sval[1][r]);
        for (int off = 1; off <= 8; off <<= 1) t = fmaxf(t, __shfl_xor(t, off, 32));
        const float mnew = fmaxf(mrow[r], t);
        corr[r] = exp2f(mrow[r] - mnew);
        mrow[r] = mnew;
        float p0, p1;
        if (edge) {
          p0 = (sval[0][r] <= -1e29f) ? 0.f : exp2f(sval[0][r] - mnew);
          p1 = (sval[1][r] <= -1e29f) ? 0.f : exp2f(sval[1][r] - mnew);
        } else {
          p0 = exp2f(sval[0][r] - mnew);
          p1 = exp2f(sval[1][r] - mnew);
        }
        sval[0][r] = p0; sval[1][r] = p1;
        float ps = p0 + p1;
        for (int off = 1; off <= 8; off <<= 1) ps += __shfl_xor(ps, off, 32);
        lrow[r] = lrow[r] * corr[r] + ps;
      }
#pragma unroll
      for (int j = 0; j < 8; ++j)
#pragma unroll
        for (int r = 0; r < 8; ++r) oacc[j][r] *= corr[r];

      // ---- relayout P (C-layout -> A-layout) via per-wave LDS ----
#pragma unroll
      for (int r = 0; r < 8; ++r) {
        Pw[wave][rowsel + r][mh]      = (bf16)sval[0][r];
        Pw[wave][rowsel + r][16 + mh] = (bf16)sval[1][r];
      }
    }
    __syncthreads();  // uniform across all waves

    if (active) {
      FragBF16 pf;
      pf.h[0] = *(const bf16x8*)&Pw[wave][mh][kh * 8];
      pf.h[1] = *(const bf16x8*)&Pw[wave][mh][16 + kh * 8];
      // ---- preload all 8 V-fragments, then 8 independent chains ----
      FragBF16 vf[8];
#pragma unroll
      for (int j = 0; j < 8; ++j) {
        const int n = j * 16 + mh;            // hd col
        vf[j].h[0] = *(const bf16x8*)&Vt[n][kh * 8];
        vf[j].h[1] = *(const bf16x8*)&Vt[n][16 + kh * 8];
      }
#pragma unroll
      for (int j = 0; j < 8; ++j)
        oacc[j] = wmma_bf16(pf.v, vf[j].v, oacc[j]);
    }
  }

  // ---- normalize + store (attn layout [B,S,nh,hd] == [M, HID]) ----
#pragma unroll
  for (int r = 0; r < 8; ++r) {
    const float inv = (lrow[r] > 0.f) ? (1.f / lrow[r]) : 0.f;
    const int qg = qbase + rowsel + r;
    bf16* op = o + ((size_t)(b * SEQ + qg)) * HID + h * HD;
#pragma unroll
    for (int j = 0; j < 8; ++j) op[j * 16 + mh] = (bf16)(oacc[j][r] * inv);
  }
}

// =====================================================================
// launch
// =====================================================================
extern "C" void kernel_launch(void* const* d_in, const int* in_sizes, int n_in,
                              void* d_out, int out_size, void* d_ws, size_t ws_size,
                              hipStream_t stream) {
  const float* hidden = (const float*)d_in[0];
  // d_in[1] = attention_mask (exactly causal; implemented analytically)
  const float* ln1  = (const float*)d_in[2];
  const float* wq   = (const float*)d_in[3];
  const float* wk   = (const float*)d_in[4];
  const float* wv   = (const float*)d_in[5];
  const float* wo   = (const float*)d_in[6];
  const float* ln2  = (const float*)d_in[7];
  const float* wfc1 = (const float*)d_in[8];
  const float* wfc2 = (const float*)d_in[9];
  float* out = (float*)d_out;

  char* ws = (char*)d_ws;
  bf16* normed = (bf16*)(ws);                       // 16 MB
  bf16* qb     = (bf16*)(ws + (16u << 20));         // 16 MB (also attn out)
  bf16* kb     = (bf16*)(ws + (32u << 20));         // 4 MB
  bf16* vb     = (bf16*)(ws + (36u << 20));         // 4 MB
  bf16* gb     = (bf16*)(ws + (40u << 20));         // 88 MB -> total 128 MB

  // 1) RMSNorm 1
  rmsnorm_kernel<<<MTOK, 256, 0, stream>>>(hidden, ln1, normed);

  // 2) Q/K/V projections (bf16 out)
  gemm_wmma_kernel<bf16, false, false><<<dim3(HID / 128, MTOK / 128), 256, 0, stream>>>(
      normed, HID, wq, qb, nullptr, MTOK, HID, HID);
  gemm_wmma_kernel<bf16, false, false><<<dim3((NKV * HD) / 128, MTOK / 128), 256, 0, stream>>>(
      normed, HID, wk, kb, nullptr, MTOK, NKV * HD, HID);
  gemm_wmma_kernel<bf16, false, false><<<dim3((NKV * HD) / 128, MTOK / 128), 256, 0, stream>>>(
      normed, HID, wv, vb, nullptr, MTOK, NKV * HD, HID);

  // 3) causal flash attention (writes into q buffer)
  attn_kernel<<<dim3(SEQ / 128, NH, BATCH), 256, 0, stream>>>(qb, kb, vb, qb);

  // 4) O-projection + residual -> d_out (fp32 h1)
  gemm_wmma_kernel<float, false, true><<<dim3(HID / 128, MTOK / 128), 256, 0, stream>>>(
      qb, HID, wo, out, hidden, MTOK, HID, HID);

  // 5) RMSNorm 2
  rmsnorm_kernel<<<MTOK, 256, 0, stream>>>(out, ln2, normed);

  // 6) fc1 -> g = [gate | up] (bf16)
  gemm_wmma_kernel<bf16, false, false><<<dim3((2 * INTER) / 128, MTOK / 128), 256, 0, stream>>>(
      normed, HID, wfc1, gb, nullptr, MTOK, 2 * INTER, HID);

  // 7) fc2 with fused swiglu A-tile + residual from/into d_out
  gemm_wmma_kernel<float, true, true><<<dim3(HID / 128, MTOK / 128), 256, 0, stream>>>(
      gb, 2 * INTER, wfc2, out, out, MTOK, HID, INTER);
}
